// AttnDecoderRNN_86998857548075
// MI455X (gfx1250) — matrix operand order repaired and checked by reading, not compiled, not run on hardware
//
#include <hip/hip_runtime.h>

// ---------------------------------------------------------------------------
// AttnDecoderRNN for MI455X (gfx1250, wave32, WMMA).
//
// All three per-step GEMMs run on v_wmma_f32_16x16x32_bf16 (weights converted
// once per call to bf16, f32 accumulate). Weights (~31 MB bf16) stay resident
// in the 192 MB L2 across the 48 serialized steps, so each step is
// L2-bandwidth bound. This version stages A/B chunks through LDS with a
// double-buffered software pipeline: a 256-thread block computes a 64x32
// macro-tile (4 M-tiles x 2 N-tiles), sharing each B fragment across 4 waves
// and each A fragment across 2 waves -> ~2.7x less L2 traffic than the
// wave-per-tile version.
// ---------------------------------------------------------------------------

typedef __bf16 bf16;
typedef __attribute__((ext_vector_type(16))) __bf16 v16bf;
typedef __attribute__((ext_vector_type(8)))  float  v8f;

#define HDIM 1024
#define BDIM 64
#define TDIM 48
#define SDIM 48
#define KC   64                      // K-chunk staged in LDS per pipeline stage
#define LDSP (KC + 8)                // padded LDS row stride (breaks bank align)
#define NEG_INF (-__builtin_inff())

__device__ __forceinline__ float sigmoidf_(float x) { return 1.0f / (1.0f + expf(-x)); }

// ---------------------------------------------------------------------------
// Prep kernels (run once per call)
// ---------------------------------------------------------------------------

// Fuse w_ih [4096,2048] and w_hh [4096,1024] into wcat [4096,3072] bf16.
__global__ void prep_wcat(const float* __restrict__ w_ih,
                          const float* __restrict__ w_hh,
                          bf16* __restrict__ wcat) {
    int idx = blockIdx.x * 256 + threadIdx.x;            // < 4096*3072
    if (idx >= 4096 * 3072) return;
    int n = idx / 3072, k = idx - n * 3072;
    float v = (k < 2048) ? w_ih[(size_t)n * 2048 + k]
                         : w_hh[(size_t)n * 1024 + (k - 2048)];
    wcat[idx] = (bf16)v;
}

__global__ void conv_f32_bf16(const float* __restrict__ src,
                              bf16* __restrict__ dst, int n) {
    int idx = blockIdx.x * 256 + threadIdx.x;
    if (idx < n) dst[idx] = (bf16)src[idx];
}

// Embedding gather for all T steps -> bf16 [T,B,H].
__global__ void prep_emb(const int* __restrict__ tokens,
                         const float* __restrict__ table,
                         bf16* __restrict__ emb) {
    int idx = blockIdx.x * 256 + threadIdx.x;            // < T*B*H
    if (idx >= TDIM * BDIM * HDIM) return;
    int tb = idx >> 10;                                  // t*B + b
    int j  = idx & (HDIM - 1);
    int tok = tokens[tb];
    emb[idx] = (bf16)table[(size_t)tok * HDIM + j];
}

// Initialize recurrent state.
__global__ void prep_state(const float* __restrict__ enc_hidden,
                           const float* __restrict__ enc_cell,
                           float* __restrict__ c,
                           bf16* __restrict__ h_bf,
                           bf16* __restrict__ feed_bf) {
    int idx = blockIdx.x * 256 + threadIdx.x;            // < B*H
    if (idx >= BDIM * HDIM) return;
    c[idx]       = enc_cell[idx];
    h_bf[idx]    = (bf16)enc_hidden[idx];
    feed_bf[idx] = (bf16)0.0f;
}

// ---------------------------------------------------------------------------
// WMMA GEMM:  C[64,N] = A[64,K] * W[N,K]^T  (+bias, optional tanh epilogue)
// A given as up to three contiguous [64,1024] bf16 segments along K.
// Block = 256 threads = 8 waves -> 64x32 macro-tile (mt = wave&3, ntl = wave>>2).
// K consumed in KC-wide chunks through double-buffered LDS; next chunk is
// fetched to registers while WMMAs consume the current one.
// ---------------------------------------------------------------------------
__global__ __launch_bounds__(256)
void wmma_gemm_bf16(const bf16* __restrict__ a0, const bf16* __restrict__ a1,
                    const bf16* __restrict__ a2,
                    const bf16* __restrict__ W, int K, int N,
                    const float* __restrict__ bias_a,
                    const float* __restrict__ bias_b,
                    float* __restrict__ C, int ldc,
                    int mode, bf16* __restrict__ out_bf) {
    __shared__ bf16 As[2][64][LDSP];                      // 2 * 64 * 72 * 2B = 18 KB
    __shared__ bf16 Bs[2][32][LDSP];                      // 2 * 32 * 72 * 2B =  9 KB

    const int tid  = threadIdx.x;
    const int lane = tid & 31;
    const int w    = tid >> 5;
    const int mt   = w & 3;                               // M-tile of this wave
    const int ntl  = w >> 2;                              // local N-tile (0/1)
    const int half = lane >> 4;                           // K-half select (ISA layout)
    const int m    = mt * 16 + (lane & 15);               // A row in macro-tile
    const int nl   = ntl * 16 + (lane & 15);              // B row (=N col) in macro-tile
    const int ng   = blockIdx.x * 32 + nl;                // global N column

    // Cooperative-load coordinates.
    const int ar = tid >> 2, ac = (tid & 3) * 16;         // A: 64 rows x 4x16 cols
    const int br = tid >> 3, bc = (tid & 7) * 8;          // B: 32 rows x 8x8  cols
    const bf16* wrow = W + (size_t)(blockIdx.x * 32 + br) * (size_t)K;

    uint4 ra0, ra1, rb0;
    auto fetch = [&](int kbase) {
        const bf16* seg = (kbase < HDIM) ? a0 : (kbase < 2 * HDIM) ? a1 : a2;
        const bf16* ap = seg + (size_t)ar * HDIM + (kbase & (HDIM - 1)) + ac;
        ra0 = *(const uint4*)(ap);
        ra1 = *(const uint4*)(ap + 8);
        rb0 = *(const uint4*)(wrow + kbase + bc);
    };
    auto stage = [&](int buf) {
        *(uint4*)(&As[buf][ar][ac])     = ra0;
        *(uint4*)(&As[buf][ar][ac] + 8) = ra1;
        *(uint4*)(&Bs[buf][br][bc])     = rb0;
    };

    fetch(0);
    stage(0);
    __syncthreads();

    v8f acc = {};
    const int nchunks = K / KC;
    for (int c = 0; c < nchunks; ++c) {
        const int buf = c & 1;
        if (c + 1 < nchunks) fetch((c + 1) * KC);         // overlap with compute

        union Frag { uint4 u[2]; v16bf v; };
#pragma unroll
        for (int kk = 0; kk < KC; kk += 32) {
            Frag fa, fb;
            // A 16x32 bf16 fragment (ISA layout): lanes 0-15 K={kk..+7, kk+16..+23},
            // lanes 16-31 K={kk+8..+15, kk+24..+31}.
            const bf16* ap = &As[buf][m][kk + half * 8];
            fa.u[0] = *(const uint4*)(ap);
            fa.u[1] = *(const uint4*)(ap + 16);
            // B 32x16 fragment: lane = column, same K split as A (lanes 0-15 K={kk..+7, kk+16..+23}, lanes 16-31 K={kk+8..+15, kk+24..+31}).
            const bf16* bp = &Bs[buf][nl][kk + half * 8];
            fb.u[0] = *(const uint4*)(bp);
            fb.u[1] = *(const uint4*)(bp + 16);
            acc = __builtin_amdgcn_wmma_f32_16x16x32_bf16(
                false, fa.v, false, fb.v, (short)0, acc, false, false);
        }
        if (c + 1 < nchunks) stage(buf ^ 1);
        __syncthreads();                                  // buf^1 ready / buf free
    }

    float bias = bias_a[ng];
    if (bias_b) bias += bias_b[ng];
#pragma unroll
    for (int r = 0; r < 8; ++r) {
        // C VGPR r: lanes 0-15 -> M=r, lanes 16-31 -> M=8+r.
        const int mg = mt * 16 + half * 8 + r;
        float v = acc[r] + bias;
        if (mode == 1) {
            v = tanhf(v);
            out_bf[(size_t)mg * HDIM + ng] = (bf16)v;     // feed for next step
        }
        C[(size_t)mg * ldc + ng] = v;
    }
}

// ---------------------------------------------------------------------------
// LSTM elementwise update: gates [64,4096] (i,f,g,o) -> c (f32), h (bf16).
// ---------------------------------------------------------------------------
__global__ void lstm_update(const float* __restrict__ gates,
                            float* __restrict__ c,
                            bf16* __restrict__ h_bf) {
    int idx = blockIdx.x * 256 + threadIdx.x;             // < B*H
    if (idx >= BDIM * HDIM) return;
    int b = idx >> 10, j = idx & (HDIM - 1);
    const float* g = gates + (size_t)b * 4 * HDIM;
    float ig = sigmoidf_(g[j]);
    float fg = sigmoidf_(g[HDIM + j]);
    float gg = tanhf(g[2 * HDIM + j]);
    float og = sigmoidf_(g[3 * HDIM + j]);
    float cn = fg * c[idx] + ig * gg;
    c[idx] = cn;
    h_bf[idx] = (bf16)(og * tanhf(cn));
}

// ---------------------------------------------------------------------------
// Attention for one step: scores = enc_outs . q, softmax over S, ctx.
// One block (256 thr, 8 waves) per batch element. Writes attn weights
// straight into d_out [B,T,S] and ctx as bf16 for the out-GEMM.
// ---------------------------------------------------------------------------
__global__ __launch_bounds__(256)
void attn_kernel(const float* __restrict__ enc_outs,
                 const float* __restrict__ q,
                 const unsigned char* __restrict__ pad,
                 float* __restrict__ attn_out,            // [B,T,S] region
                 bf16* __restrict__ ctx_bf, int t) {
    __shared__ float sc[SDIM];
    const int b = blockIdx.x;
    const int tid = threadIdx.x;
    const int lane = tid & 31, warp = tid >> 5;
    const float* qb = q + (size_t)b * HDIM;

    for (int s = warp; s < SDIM; s += 8) {
        const float* e = enc_outs + ((size_t)s * BDIM + b) * HDIM;
        float p = 0.f;
        for (int j = lane; j < HDIM; j += 32) p += e[j] * qb[j];
#pragma unroll
        for (int o = 16; o > 0; o >>= 1) p += __shfl_xor(p, o, 32);
        if (lane == 0) sc[s] = pad[s * BDIM + b] ? NEG_INF : p;
    }
    __syncthreads();
    float mx = NEG_INF;
    for (int s = 0; s < SDIM; ++s) mx = fmaxf(mx, sc[s]);
    float sum = 0.f;
    for (int s = 0; s < SDIM; ++s) sum += expf(sc[s] - mx);
    const float inv = 1.0f / sum;
    __syncthreads();
    if (tid < SDIM) {
        float a = expf(sc[tid] - mx) * inv;
        sc[tid] = a;
        attn_out[(size_t)b * TDIM * SDIM + t * SDIM + tid] = a;
    }
    __syncthreads();
    for (int j = tid; j < HDIM; j += 256) {
        float acc = 0.f;
#pragma unroll 4
        for (int s = 0; s < SDIM; ++s)
            acc += sc[s] * enc_outs[((size_t)s * BDIM + b) * HDIM + j];
        ctx_bf[(size_t)b * HDIM + j] = (bf16)acc;
    }
}

// ---------------------------------------------------------------------------
// Finalization
// ---------------------------------------------------------------------------

// last_out[b,h] = raw out[b, T-1, h] (copied BEFORE in-place log-softmax).
__global__ void copy_last(const float* __restrict__ x, float* __restrict__ last) {
    int idx = blockIdx.x * 256 + threadIdx.x;             // < B*H
    if (idx >= BDIM * HDIM) return;
    int b = idx >> 10, h = idx & (HDIM - 1);
    last[idx] = x[(size_t)b * TDIM * HDIM + (TDIM - 1) * HDIM + h];
}

// In-place log-softmax over the batch axis of x[B,T,H] (legacy dim=0).
// One thread owns one (t,h) column of 64 elements.
__global__ void logsoftmax_b(float* __restrict__ x) {
    int idx = blockIdx.x * 256 + threadIdx.x;             // < T*H
    if (idx >= TDIM * HDIM) return;
    float mx = NEG_INF;
    for (int b = 0; b < BDIM; ++b)
        mx = fmaxf(mx, x[(size_t)b * TDIM * HDIM + idx]);
    float sum = 0.f;
    for (int b = 0; b < BDIM; ++b)
        sum += expf(x[(size_t)b * TDIM * HDIM + idx] - mx);
    float lse = mx + logf(sum);
    for (int b = 0; b < BDIM; ++b)
        x[(size_t)b * TDIM * HDIM + idx] -= lse;
}

// ---------------------------------------------------------------------------
// Host launcher
// ---------------------------------------------------------------------------
extern "C" void kernel_launch(void* const* d_in, const int* in_sizes, int n_in,
                              void* d_out, int out_size, void* d_ws, size_t ws_size,
                              hipStream_t stream) {
    const int*   tokens     = (const int*)d_in[0];        // [T,B]
    const float* enc_outs   = (const float*)d_in[1];      // [S,B,H]
    const float* enc_hidden = (const float*)d_in[2];      // [B,H]
    const float* enc_cell   = (const float*)d_in[3];      // [B,H]
    const unsigned char* pad = (const unsigned char*)d_in[4]; // [S,B] bool (all 0)
    const float* emb_table  = (const float*)d_in[5];      // [V,H]
    const float* w_ih  = (const float*)d_in[6];           // [4H,2H]
    const float* w_hh  = (const float*)d_in[7];           // [4H,H]
    const float* b_ih  = (const float*)d_in[8];           // [4H]
    const float* b_hh  = (const float*)d_in[9];           // [4H]
    const float* w_inp = (const float*)d_in[10];          // [H,H]
    const float* b_inp = (const float*)d_in[11];          // [H]
    const float* w_outp = (const float*)d_in[12];         // [H,2H]
    const float* b_outp = (const float*)d_in[13];         // [H]

    float* out = (float*)d_out;
    float* outsBTH = out;                                 // [B,T,H] (raw, then logsoftmax'd)
    float* last    = out + (size_t)BDIM * TDIM * HDIM;    // [B,H]
    float* attnO   = last + (size_t)BDIM * HDIM;          // [B,T,S]

    // Workspace carve (~40 MB), 256B aligned slices.
    char* p = (char*)d_ws;
    auto take = [&](size_t bytes) -> char* {
        char* r = p;
        p += (bytes + 255) & ~(size_t)255;
        return r;
    };
    bf16*  wcat    = (bf16*)take((size_t)4096 * 3072 * 2);   // fused w_ih|w_hh
    bf16*  winp_bf = (bf16*)take((size_t)HDIM * HDIM * 2);
    bf16*  woutp_bf= (bf16*)take((size_t)HDIM * 2 * HDIM * 2);
    bf16*  emb_all = (bf16*)take((size_t)TDIM * BDIM * HDIM * 2);
    bf16*  feed_bf = (bf16*)take((size_t)BDIM * HDIM * 2);
    bf16*  h_bf    = (bf16*)take((size_t)BDIM * HDIM * 2);
    bf16*  ctx_bf  = (bf16*)take((size_t)BDIM * HDIM * 2);
    float* c_st    = (float*)take((size_t)BDIM * HDIM * 4);
    float* q_st    = (float*)take((size_t)BDIM * HDIM * 4);
    float* gates   = (float*)take((size_t)BDIM * 4 * HDIM * 4);

    // ---- prep (once per call; deterministic) ----
    prep_wcat<<<(4096 * 3072 + 255) / 256, 256, 0, stream>>>(w_ih, w_hh, wcat);
    conv_f32_bf16<<<(HDIM * HDIM + 255) / 256, 256, 0, stream>>>(w_inp, winp_bf, HDIM * HDIM);
    conv_f32_bf16<<<(HDIM * 2 * HDIM + 255) / 256, 256, 0, stream>>>(w_outp, woutp_bf, HDIM * 2 * HDIM);
    prep_emb<<<(TDIM * BDIM * HDIM + 255) / 256, 256, 0, stream>>>(tokens, emb_table, emb_all);
    prep_state<<<(BDIM * HDIM + 255) / 256, 256, 0, stream>>>(enc_hidden, enc_cell, c_st, h_bf, feed_bf);

    // ---- 48 serialized decoder steps ----
    for (int t = 0; t < TDIM; ++t) {
        // gates[64,4096] = [emb_t|feed|h] @ wcat^T + b_ih + b_hh
        wmma_gemm_bf16<<<4096 / 32, 256, 0, stream>>>(
            emb_all + (size_t)t * BDIM * HDIM, feed_bf, h_bf,
            wcat, 3072, 4096, b_ih, b_hh, gates, 4096, 0, nullptr);
        // LSTM cell -> c (f32), h (bf16)
        lstm_update<<<(BDIM * HDIM + 255) / 256, 256, 0, stream>>>(gates, c_st, h_bf);
        // q[64,1024] = h @ w_inp^T + b_inp
        wmma_gemm_bf16<<<1024 / 32, 256, 0, stream>>>(
            h_bf, h_bf, h_bf, winp_bf, 1024, 1024, b_inp, nullptr,
            q_st, 1024, 0, nullptr);
        // attention: softmax(enc.q) over S, ctx; attn -> d_out [B,T,S]
        attn_kernel<<<BDIM, 256, 0, stream>>>(enc_outs, q_st, pad, attnO, ctx_bf, t);
        // out[64,1024] = tanh([ctx|h] @ w_outp^T + b_outp)
        //   -> d_out[b, t, :] (transposed layout) and feed_bf for step t+1
        wmma_gemm_bf16<<<1024 / 32, 256, 0, stream>>>(
            ctx_bf, h_bf, h_bf, woutp_bf, 2048, 1024, b_outp, nullptr,
            outsBTH + (size_t)t * HDIM, TDIM * HDIM, 1, feed_bf);
    }

    // ---- finalize: last_out copy first, then in-place batch log-softmax ----
    copy_last<<<(BDIM * HDIM + 255) / 256, 256, 0, stream>>>(outsBTH, last);
    logsoftmax_b<<<(TDIM * HDIM + 255) / 256, 256, 0, stream>>>(outsBTH);
}
